// CustomMultiHeadAttention_25082609009124
// MI455X (gfx1250) — matrix operand order repaired and checked by reading, not compiled, not run on hardware
//
#include <hip/hip_runtime.h>
#include <hip/hip_bf16.h>

// ---------------------------------------------------------------------------
// MI455X (gfx1250) multi-head attention, bf16 WMMA everywhere, f32 accumulate.
// B=2, S=2048, D=2048, H=16, Hd=128.  ~206 GFLOP, ~200MB min HBM traffic
// (~9us @ 23.3TB/s) => compute bound => all matmuls via v_wmma_f32_16x16x32_bf16.
// One bandwidth-cheap pre-pass converts weights AND activations to bf16, so
// every GEMM inner loop is pure {global_load_b128 x12 -> v_wmma x8} per K-step.
// V projection written pre-transposed [B,H,Hd,S] so P.V B-fragments are
// contiguous b128 loads; softmax runs in exp2 domain (scale folded into Q).
// ---------------------------------------------------------------------------

typedef __bf16 bf16_t;
typedef bf16_t v16bf __attribute__((ext_vector_type(16)));
typedef bf16_t v8bf  __attribute__((ext_vector_type(8)));
typedef float  v8f   __attribute__((ext_vector_type(8)));

#define EMBED  2048
#define SEQ    2048
#define BATCH  2
#define HEADS  16
#define HD     128
#define MROWS  (BATCH*SEQ)   // 4096

static __device__ inline v8f wmma_bf16(v16bf a, v16bf b, v8f c) {
  // (neg_a, A, neg_b, B, c_mod, C, reuse_a, reuse_b)
  return __builtin_amdgcn_wmma_f32_16x16x32_bf16(false, a, false, b, (short)0, c,
                                                 false, false);
}

static __device__ inline v8f v8f_zero() {
  v8f z;
#pragma unroll
  for (int i = 0; i < 8; ++i) z[i] = 0.0f;
  return z;
}

static __device__ inline v16bf cat8(v8bf lo, v8bf hi) {
  v16bf r;
#pragma unroll
  for (int i = 0; i < 8; ++i) { r[i] = lo[i]; r[8 + i] = hi[i]; }
  return r;
}

// A-fragment (16x32 bf16), row = lane&15.
// Lane half 0: K = [0..7],[16..23]; half 1: K = [8..15],[24..31].
// p points at (row, k0 + hf*8): two b128 loads at p and p+16.
static __device__ inline v16bf a_frag_from_bf16(const bf16_t* __restrict__ p) {
  return cat8(*(const v8bf*)p, *(const v8bf*)(p + 16));
}

// B-fragment (32x16 bf16): col = lane&15, same K order as A-fragment
// (half 0: K [0..7],[16..23]; half 1: [8..15],[24..31]); p at k0 + hf*8.
static __device__ inline v16bf b_frag_from_bf16(const bf16_t* __restrict__ p) {
  return cat8(*(const v8bf*)p, *(const v8bf*)(p + 16));
}

static __device__ inline float redmax16(float v) {
  v = fmaxf(v, __shfl_xor(v, 1, 32));
  v = fmaxf(v, __shfl_xor(v, 2, 32));
  v = fmaxf(v, __shfl_xor(v, 4, 32));
  v = fmaxf(v, __shfl_xor(v, 8, 32));
  return v;
}
static __device__ inline float redsum16(float v) {
  v += __shfl_xor(v, 1, 32);
  v += __shfl_xor(v, 2, 32);
  v += __shfl_xor(v, 4, 32);
  v += __shfl_xor(v, 8, 32);
  return v;
}

// ---------------------------------------------------------------------------
// Elementwise f32 -> bf16 pre-pass (weights + activations; ~10us total HBM)
// ---------------------------------------------------------------------------
__global__ __launch_bounds__(256) void cvt_f32_bf16(const float* __restrict__ src,
                                                    bf16_t* __restrict__ dst,
                                                    int n8) {
  const int i = blockIdx.x * blockDim.x + threadIdx.x;
  if (i >= n8) return;
  const float4* s = (const float4*)src + (size_t)i * 2;
  float4 f0 = s[0], f1 = s[1];
  v8bf r;
  r[0]=(bf16_t)f0.x; r[1]=(bf16_t)f0.y; r[2]=(bf16_t)f0.z; r[3]=(bf16_t)f0.w;
  r[4]=(bf16_t)f1.x; r[5]=(bf16_t)f1.y; r[6]=(bf16_t)f1.z; r[7]=(bf16_t)f1.w;
  *((v8bf*)dst + i) = r;
}

// ---------------------------------------------------------------------------
// GEMM: out[m,n] = scale * (X[m,:] . W[n,:] + bias[n])
// X: [MROWS, EMBED] bf16 row-major, W: [EMBED, EMBED] bf16 row-major.
// OUT_KIND 0: bf16 -> [B,H,S,Hd]   (Q,K head-split layout)
// OUT_KIND 1: f32  -> [MROWS,EMBED] (final output)
// OUT_KIND 2: bf16 -> [B,H,Hd,S]   (V head-split, pre-transposed for P.V)
// Block = 256 threads (8 waves): 128x128 tile; wave tile 32x64; K-step 32.
// Inner loop: 12 b128 loads -> 8 WMMAs, zero conversion VALU.
// ---------------------------------------------------------------------------
template <int OUT_KIND>
__global__ __launch_bounds__(256) void gemm_xwt(const bf16_t* __restrict__ X,
                                                const bf16_t* __restrict__ W,
                                                const float* __restrict__ bias,
                                                void* __restrict__ outp,
                                                float scale) {
  const int K    = EMBED;
  const int lane = threadIdx.x & 31;
  const int wave = threadIdx.x >> 5;
  const int hf   = lane >> 4;   // lane half (wave32)
  const int lr   = lane & 15;
  const int m0 = blockIdx.x * 128 + (wave >> 1) * 32;
  const int n0 = blockIdx.y * 128 + (wave & 1) * 64;

  v8f acc[2][4];
#pragma unroll
  for (int i = 0; i < 2; ++i)
#pragma unroll
    for (int j = 0; j < 4; ++j) acc[i][j] = v8f_zero();

  for (int k0 = 0; k0 < K; k0 += 32) {
    v16bf a[2];
#pragma unroll
    for (int i = 0; i < 2; ++i)
      a[i] = a_frag_from_bf16(X + (size_t)(m0 + i * 16 + lr) * K + k0 + hf * 8);
    v16bf b[4];
#pragma unroll
    for (int j = 0; j < 4; ++j)
      b[j] = b_frag_from_bf16(W + (size_t)(n0 + j * 16 + lr) * K + k0 + hf * 8);
#pragma unroll
    for (int i = 0; i < 2; ++i)
#pragma unroll
      for (int j = 0; j < 4; ++j) acc[i][j] = wmma_bf16(a[i], b[j], acc[i][j]);
  }

#pragma unroll
  for (int i = 0; i < 2; ++i)
#pragma unroll
    for (int j = 0; j < 4; ++j) {
      const int col = n0 + j * 16 + lr;
      const float bv = bias[col];
      const int h = col >> 7, d = col & (HD - 1);
#pragma unroll
      for (int r = 0; r < 8; ++r) {
        const int row = m0 + i * 16 + hf * 8 + r;
        const int bb = row >> 11, s = row & (SEQ - 1);
        const float v = (acc[i][j][r] + bv) * scale;
        if (OUT_KIND == 0) {
          ((bf16_t*)outp)[(((size_t)bb * HEADS + h) * SEQ + s) * HD + d] =
              (bf16_t)v;
        } else if (OUT_KIND == 1) {
          ((float*)outp)[(size_t)row * EMBED + col] = v;
        } else {  // V: [B,H,Hd,S]
          ((bf16_t*)outp)[(((size_t)bb * HEADS + h) * HD + d) * SEQ + s] =
              (bf16_t)v;
        }
      }
    }
}

// ---------------------------------------------------------------------------
// Flash attention. Q,K bf16 [B,H,S,Hd]; V bf16 [B,H,Hd,S] (pre-transposed);
// output bf16 [B,S,D].  Block = 256 threads (8 waves), each wave independent
// (no barriers): 16 query rows per wave, key tile = 32.
// Per key tile / wave: 8 score WMMAs + 8 PV WMMAs.  Q already carries
// log2(e)/sqrt(Hd), so softmax is pure exp2 (single v_exp_f32 per element).
// ---------------------------------------------------------------------------
__global__ __launch_bounds__(256) void flash_attn(const bf16_t* __restrict__ Qh,
                                                  const bf16_t* __restrict__ Kh,
                                                  const bf16_t* __restrict__ Vh,
                                                  bf16_t* __restrict__ O) {
  __shared__ __align__(16) bf16_t Plds[8][16][40];  // per-wave P transpose pad

  const int lane = threadIdx.x & 31;
  const int wave = threadIdx.x >> 5;
  const int hf = lane >> 4, lr = lane & 15;
  const int b = blockIdx.z, h = blockIdx.y;
  const int s0 = blockIdx.x * 128 + wave * 16;  // this wave's 16 query rows

  const size_t head = ((size_t)b * HEADS + h) * SEQ * HD;
  const bf16_t* Q = Qh + head;   // [S][Hd]
  const bf16_t* Kp = Kh + head;  // [S][Hd]
  const bf16_t* Vp = Vh + head;  // [Hd][S]  (transposed)

  // Preload Q A-fragments along Hd (4 K-steps of 32), reused all key blocks.
  v16bf qf[4];
#pragma unroll
  for (int kq = 0; kq < 4; ++kq)
    qf[kq] = a_frag_from_bf16(Q + (size_t)(s0 + lr) * HD + kq * 32 + hf * 8);

  v8f o[8];
#pragma unroll
  for (int j = 0; j < 8; ++j) o[j] = v8f_zero();
  float mi[8], li[8];
#pragma unroll
  for (int r = 0; r < 8; ++r) { mi[r] = -1e30f; li[r] = 0.0f; }

  for (int t0 = 0; t0 < SEQ; t0 += 32) {
    // ---- scores (log2-domain) for this wave's 16x32 tile ----
    v8f sf[2];
    sf[0] = v8f_zero(); sf[1] = v8f_zero();
#pragma unroll
    for (int kq = 0; kq < 4; ++kq)
#pragma unroll
      for (int j = 0; j < 2; ++j) {
        v16bf kb = b_frag_from_bf16(Kp + (size_t)(t0 + j * 16 + lr) * HD +
                                    kq * 32 + hf * 8);
        sf[j] = wmma_bf16(qf[kq], kb, sf[j]);
      }

    // ---- online softmax, exp2 domain (rows across 16-lane half-groups) ----
    float corr[8];
#pragma unroll
    for (int r = 0; r < 8; ++r) {
      float mx = redmax16(fmaxf(sf[0][r], sf[1][r]));
      float mn = fmaxf(mi[r], mx);
      float c = exp2f(mi[r] - mn);
      float p0 = exp2f(sf[0][r] - mn);
      float p1 = exp2f(sf[1][r] - mn);
      sf[0][r] = p0; sf[1][r] = p1;
      li[r] = li[r] * c + redsum16(p0 + p1);
      mi[r] = mn;
      corr[r] = c;
    }
#pragma unroll
    for (int j = 0; j < 8; ++j)
#pragma unroll
      for (int r = 0; r < 8; ++r) o[j][r] *= corr[r];

    // ---- transpose P through per-wave LDS into A-fragment layout ----
    // (wave-private scratch: no barrier, DS counters order store->load)
#pragma unroll
    for (int j = 0; j < 2; ++j)
#pragma unroll
      for (int r = 0; r < 8; ++r)
        Plds[wave][hf * 8 + r][j * 16 + lr] = (bf16_t)sf[j][r];
    const bf16_t* pr = &Plds[wave][lr][0];
    v16bf pa = cat8(*(const v8bf*)(pr + hf * 8),
                    *(const v8bf*)(pr + 16 + hf * 8));

    // ---- O += P (16x32) x V (32x128): B-frags direct from [Hd][S] ----
#pragma unroll
    for (int j = 0; j < 8; ++j) {
      v16bf vb = b_frag_from_bf16(Vp + (size_t)(j * 16 + lr) * SEQ + t0 +
                                  hf * 8);
      o[j] = wmma_bf16(pa, vb, o[j]);
    }
  }

  // ---- finalize: 8 reciprocals + 64 multiplies, write bf16 [B,S,D] ----
  float inv[8];
#pragma unroll
  for (int r = 0; r < 8; ++r) inv[r] = 1.0f / li[r];
#pragma unroll
  for (int j = 0; j < 8; ++j)
#pragma unroll
    for (int r = 0; r < 8; ++r) {
      const int row = s0 + hf * 8 + r;
      const int col = h * HD + j * 16 + lr;
      O[((size_t)b * SEQ + row) * EMBED + col] = (bf16_t)(o[j][r] * inv[r]);
    }
}

// ---------------------------------------------------------------------------
extern "C" void kernel_launch(void* const* d_in, const int* in_sizes, int n_in,
                              void* d_out, int out_size, void* d_ws,
                              size_t ws_size, hipStream_t stream) {
  const float* query  = (const float*)d_in[0];
  const float* key_in = (const float*)d_in[1];
  const float* value  = (const float*)d_in[2];
  const float* Wq = (const float*)d_in[3];
  const float* bq = (const float*)d_in[4];
  const float* Wk = (const float*)d_in[5];
  const float* bk = (const float*)d_in[6];
  const float* Wv = (const float*)d_in[7];
  const float* bv = (const float*)d_in[8];
  const float* Wo = (const float*)d_in[9];
  const float* bo = (const float*)d_in[10];

  char* ws = (char*)d_ws;
  const size_t MB = (size_t)1 << 20;
  bf16_t* Wqb = (bf16_t*)(ws);              // 8 MB  bf16 [D,D]
  bf16_t* Wkb = (bf16_t*)(ws + 8 * MB);     // 8 MB
  bf16_t* Wvb = (bf16_t*)(ws + 16 * MB);    // 8 MB
  bf16_t* Wob = (bf16_t*)(ws + 24 * MB);    // 8 MB
  bf16_t* Xq  = (bf16_t*)(ws + 32 * MB);    // 16 MB bf16 [MROWS,D]
  bf16_t* Xk  = (bf16_t*)(ws + 48 * MB);    // 16 MB
  bf16_t* Xv  = (bf16_t*)(ws + 64 * MB);    // 16 MB
  bf16_t* Qh  = (bf16_t*)(ws + 80 * MB);    // 16 MB [B,H,S,Hd]
  bf16_t* Kh  = (bf16_t*)(ws + 96 * MB);    // 16 MB [B,H,S,Hd]
  bf16_t* Vh  = (bf16_t*)(ws + 112 * MB);   // 16 MB [B,H,Hd,S]
  bf16_t* Oa  = (bf16_t*)(ws + 128 * MB);   // 16 MB [B,S,D]

  const int wn8 = (EMBED * EMBED) / 8;   // 524288
  const int an8 = (MROWS * EMBED) / 8;   // 1048576
  dim3 cb(256);
  hipLaunchKernelGGL(cvt_f32_bf16, dim3(wn8 / 256), cb, 0, stream, Wq, Wqb, wn8);
  hipLaunchKernelGGL(cvt_f32_bf16, dim3(wn8 / 256), cb, 0, stream, Wk, Wkb, wn8);
  hipLaunchKernelGGL(cvt_f32_bf16, dim3(wn8 / 256), cb, 0, stream, Wv, Wvb, wn8);
  hipLaunchKernelGGL(cvt_f32_bf16, dim3(wn8 / 256), cb, 0, stream, Wo, Wob, wn8);
  hipLaunchKernelGGL(cvt_f32_bf16, dim3(an8 / 256), cb, 0, stream, query, Xq, an8);
  hipLaunchKernelGGL(cvt_f32_bf16, dim3(an8 / 256), cb, 0, stream, key_in, Xk, an8);
  hipLaunchKernelGGL(cvt_f32_bf16, dim3(an8 / 256), cb, 0, stream, value, Xv, an8);

  dim3 gg(MROWS / 128, EMBED / 128);  // 32 x 16
  dim3 bk256(256);
  // Q scale: log2(e)/sqrt(Hd) so flash softmax runs in pure exp2 domain.
  const float qscale = 0.12751744951379378f;

  hipLaunchKernelGGL((gemm_xwt<0>), gg, bk256, 0, stream, Xq, Wqb, bq,
                     (void*)Qh, qscale);
  hipLaunchKernelGGL((gemm_xwt<0>), gg, bk256, 0, stream, Xk, Wkb, bk,
                     (void*)Kh, 1.0f);
  hipLaunchKernelGGL((gemm_xwt<2>), gg, bk256, 0, stream, Xv, Wvb, bv,
                     (void*)Vh, 1.0f);
  hipLaunchKernelGGL(flash_attn, dim3(SEQ / 128, HEADS, BATCH), bk256, 0,
                     stream, Qh, Kh, Vh, Oa);
  hipLaunchKernelGGL((gemm_xwt<1>), gg, bk256, 0, stream, Oa, Wob, bo, d_out,
                     1.0f);
}
